// Had_linear_86930138071526
// MI455X (gfx1250) — 8-bit matrix operand order repaired and checked by reading, compiled, not run on hardware
//
#include <hip/hip_runtime.h>

typedef int   v16i __attribute__((ext_vector_type(16)));
typedef float v8f  __attribute__((ext_vector_type(8)));

#define M_ROWS 8192
#define N_COLS 4096
#define K_DIM  4096
#define BM 128
#define BN 128
#define BK 128
#define LDSS 144

#if __has_builtin(__builtin_amdgcn_global_load_async_to_lds_b128)
#define USE_ASYNC_LDS 1
#else
#define USE_ASYNC_LDS 0
#endif

typedef int v4i_vs __attribute__((vector_size(4 * sizeof(int))));
typedef __attribute__((address_space(1))) v4i_vs glb_v4i;
typedef __attribute__((address_space(3))) v4i_vs lds_v4i;

__device__ __forceinline__ glb_v4i* as_global(const void* p) {
    return (glb_v4i*)(unsigned long long)p;
}
__device__ __forceinline__ lds_v4i* as_lds(const void* p) {
    return (lds_v4i*)(unsigned int)(unsigned long long)p;
}

#define FP8LUT_LO 0xB8C0C4C8CACCCED0ULL
#define FP8LUT_HI 0x4E4C4A4844403800ULL

__device__ __forceinline__ unsigned int fp8_of_code(int q) {
    const unsigned int idx = (unsigned int)(q + 8);
    const unsigned long long tab = (idx & 8u) ? FP8LUT_HI : FP8LUT_LO;
    return (unsigned int)((tab >> ((idx & 7u) * 8u)) & 0xffu);
}

__global__ __launch_bounds__(256) void fwht_actquant_kernel(
    const float* __restrict__ x,
    const float* __restrict__ alpha_a, const float* __restrict__ beta_a,
    unsigned char* __restrict__ qx, float* __restrict__ sx)
{
    __shared__ float buf[4096];
    __shared__ int ssum;
    const int row = blockIdx.x;
    const int t = threadIdx.x;
    const float* xr = x + (size_t)row * K_DIM;

    for (int i = t; i < 1024; i += 256) {
        float4 v = ((const float4*)xr)[i];
        buf[i*4+0] = v.x; buf[i*4+1] = v.y; buf[i*4+2] = v.z; buf[i*4+3] = v.w;
    }
    if (t == 0) ssum = 0;
    __syncthreads();

    for (int len = 1; len < 4096; len <<= 1) {
        for (int p = t; p < 2048; p += 256) {
            int i = ((p & ~(len - 1)) << 1) | (p & (len - 1));
            int j = i + len;
            float a = buf[i], b = buf[j];
            buf[i] = a + b;
            buf[j] = a - b;
        }
        __syncthreads();
    }

    const float alpha = alpha_a[0];
    const float beta  = beta_a[0];
    const float inv_a = 1.0f / alpha;
    const float nrm   = 1.0f / 64.0f;

    int lsum = 0;
    for (int i4 = t; i4 < 1024; i4 += 256) {
        unsigned int packed = 0;
        #pragma unroll
        for (int e = 0; e < 4; ++e) {
            float v = buf[i4*4 + e] * nrm + beta;
            float qf = rintf(fminf(fmaxf(v * inv_a, -8.0f), 7.0f));
            int qi = (int)qf;
            lsum += qi;
            packed |= fp8_of_code(qi) << (8 * e);
        }
        ((unsigned int*)qx)[(size_t)row * 1024 + i4] = packed;
    }
    atomicAdd(&ssum, lsum);
    __syncthreads();
    if (t == 0) sx[row] = (float)ssum;
}

__global__ __launch_bounds__(256) void wquant_kernel(
    const float* __restrict__ w,
    const float* __restrict__ alpha_w, const float* __restrict__ beta_w,
    const float* __restrict__ a_w,
    unsigned char* __restrict__ qw, float* __restrict__ sw)
{
    __shared__ int ssum;
    const int o = blockIdx.x;
    const int t = threadIdx.x;

    float thr[15];
    #pragma unroll
    for (int i = 0; i < 15; ++i) thr[i] = 0.5f * (a_w[i] + a_w[i + 1]);

    const float alpha = alpha_w[0];
    const float beta  = beta_w[0];
    const float inv_a = 1.0f / alpha;
    const float* wr = w + (size_t)o * K_DIM;

    if (t == 0) ssum = 0;
    __syncthreads();

    int lsum = 0;
    for (int i4 = t; i4 < 1024; i4 += 256) {
        float4 v4 = ((const float4*)wr)[i4];
        float vv[4] = {v4.x, v4.y, v4.z, v4.w};
        unsigned int packed = 0;
        #pragma unroll
        for (int e = 0; e < 4; ++e) {
            float xc = fminf(fmaxf((vv[e] + beta) * inv_a, -8.0f), 7.0f);
            int q = -8;
            #pragma unroll
            for (int i = 0; i < 15; ++i) q = (xc > thr[i]) ? (i - 7) : q;
            lsum += q;
            packed |= fp8_of_code(q) << (8 * e);
        }
        ((unsigned int*)qw)[(size_t)o * 1024 + i4] = packed;
    }
    atomicAdd(&ssum, lsum);
    __syncthreads();
    if (t == 0) sw[o] = (float)ssum;
}

__global__ __launch_bounds__(512, 1) void gemm_fp8_kernel(
    const unsigned char* __restrict__ qx, const unsigned char* __restrict__ qw,
    const float* __restrict__ bias,
    const float* __restrict__ sx, const float* __restrict__ sw,
    const float* __restrict__ alpha_a, const float* __restrict__ beta_a,
    const float* __restrict__ alpha_w, const float* __restrict__ beta_w,
    float* __restrict__ out)
{
    __shared__ __align__(16) unsigned char ldsA[2][BM * LDSS];
    __shared__ __align__(16) unsigned char ldsB[2][BN * LDSS];

    const int t    = threadIdx.x;
    const int wave = t >> 5;
    const int lane = t & 31;
    const int wm   = wave & 3;
    const int wn   = wave >> 2;
    const int l16  = lane & 15;
    const int lh   = lane >> 4;

    const int mBase = blockIdx.y * BM;
    const int nBase = blockIdx.x * BN;

    const int rowS = t >> 3;
    const int colS = (t & 7) * 16;

    v8f acc[2][2];
    #pragma unroll
    for (int fm = 0; fm < 2; ++fm)
        #pragma unroll
        for (int fn = 0; fn < 2; ++fn)
            acc[fm][fn] = (v8f)0.0f;

    int cur = 0;
    const int KT = K_DIM / BK;

#if USE_ASYNC_LDS
    #pragma unroll
    for (int c = 0; c < 2; ++c) {
        const int r = rowS + c * 64;
        __builtin_amdgcn_global_load_async_to_lds_b128(
            as_global(qx + (size_t)(mBase + r) * K_DIM + colS),
            as_lds(&ldsA[0][r * LDSS + colS]), 0, 0);
        __builtin_amdgcn_global_load_async_to_lds_b128(
            as_global(qw + (size_t)(nBase + r) * K_DIM + colS),
            as_lds(&ldsB[0][r * LDSS + colS]), 0, 0);
    }
    asm volatile("s_wait_asynccnt 0" ::: "memory");
    __syncthreads();

    for (int kt = 0; kt < KT; ++kt) {
        if (kt + 1 < KT) {
            const int nxt = cur ^ 1;
            const int k0n = (kt + 1) * BK;
            #pragma unroll
            for (int c = 0; c < 2; ++c) {
                const int r = rowS + c * 64;
                __builtin_amdgcn_global_load_async_to_lds_b128(
                    as_global(qx + (size_t)(mBase + r) * K_DIM + k0n + colS),
                    as_lds(&ldsA[nxt][r * LDSS + colS]), 0, 0);
                __builtin_amdgcn_global_load_async_to_lds_b128(
                    as_global(qw + (size_t)(nBase + r) * K_DIM + k0n + colS),
                    as_lds(&ldsB[nxt][r * LDSS + colS]), 0, 0);
            }
        }

        v16i afrag[2];
        #pragma unroll
        for (int fm = 0; fm < 2; ++fm) {
            const unsigned char* p =
                &ldsA[cur][(wm * 32 + fm * 16 + l16) * LDSS + lh * 16];
            v16i a;
            #pragma unroll
            for (int h = 0; h < 8; ++h) {
                int2 d = *(const int2*)(p + (h >> 1) * 32 + (h & 1) * 8);
                a[2 * h]     = d.x;
                a[2 * h + 1] = d.y;
            }
            afrag[fm] = a;
        }
        #pragma unroll
        for (int fn = 0; fn < 2; ++fn) {
            const unsigned char* p =
                &ldsB[cur][(wn * 32 + fn * 16 + l16) * LDSS + lh * 16];
            v16i b;
            #pragma unroll
            for (int g = 0; g < 4; ++g) {
                int4 e = *(const int4*)(p + g * 32);
                b[4 * g + 0] = e.x; b[4 * g + 1] = e.y;
                b[4 * g + 2] = e.z; b[4 * g + 3] = e.w;
            }
            #pragma unroll
            for (int fm = 0; fm < 2; ++fm) {
                acc[fm][fn] = __builtin_amdgcn_wmma_f32_16x16x128_fp8_fp8(
                    afrag[fm], b, (short)0, acc[fm][fn],
                     false,  false);
            }
        }

        asm volatile("s_wait_asynccnt 0" ::: "memory");
        __syncthreads();
        cur ^= 1;
    }
#else
    #pragma unroll
    for (int c = 0; c < 2; ++c) {
        const int r = rowS + c * 64;
        int4 ra = *(const int4*)(qx + (size_t)(mBase + r) * K_DIM + colS);
        int4 rb = *(const int4*)(qw + (size_t)(nBase + r) * K_DIM + colS);
        *(int4*)(&ldsA[0][r * LDSS + colS]) = ra;
        *(int4*)(&ldsB[0][r * LDSS + colS]) = rb;
    }
    __syncthreads();

    for (int kt = 0; kt < KT; ++kt) {
        const bool has_next = (kt + 1) < KT;
        int4 ra[2], rb[2];
        if (has_next) {
            const int k0n = (kt + 1) * BK;
            #pragma unroll
            for (int c = 0; c < 2; ++c) {
                const int r = rowS + c * 64;
                ra[c] = *(const int4*)(qx + (size_t)(mBase + r) * K_DIM + k0n + colS);
                rb[c] = *(const int4*)(qw + (size_t)(nBase + r) * K_DIM + k0n + colS);
            }
        }

        v16i afrag[2];
        #pragma unroll
        for (int fm = 0; fm < 2; ++fm) {
            const unsigned char* p =
                &ldsA[cur][(wm * 32 + fm * 16 + l16) * LDSS + lh * 16];
            v16i a;
            #pragma unroll
            for (int h = 0; h < 8; ++h) {
                int2 d = *(const int2*)(p + (h >> 1) * 32 + (h & 1) * 8);
                a[2 * h]     = d.x;
                a[2 * h + 1] = d.y;
            }
            afrag[fm] = a;
        }
        #pragma unroll
        for (int fn = 0; fn < 2; ++fn) {
            const unsigned char* p =
                &ldsB[cur][(wn * 32 + fn * 16 + l16) * LDSS + lh * 16];
            v16i b;
            #pragma unroll
            for (int g = 0; g < 4; ++g) {
                int4 e = *(const int4*)(p + g * 32);
                b[4 * g + 0] = e.x; b[4 * g + 1] = e.y;
                b[4 * g + 2] = e.z; b[4 * g + 3] = e.w;
            }
            #pragma unroll
            for (int fm = 0; fm < 2; ++fm) {
                acc[fm][fn] = __builtin_amdgcn_wmma_f32_16x16x128_fp8_fp8(
                    afrag[fm], b, (short)0, acc[fm][fn], false, false);
            }
        }

        __syncthreads();
        if (has_next) {
            const int nxt = cur ^ 1;
            #pragma unroll
            for (int c = 0; c < 2; ++c) {
                const int r = rowS + c * 64;
                *(int4*)(&ldsA[nxt][r * LDSS + colS]) = ra[c];
                *(int4*)(&ldsB[nxt][r * LDSS + colS]) = rb[c];
            }
        }
        __syncthreads();
        cur ^= 1;
    }
#endif

    const float aa = alpha_a[0], ba = beta_a[0];
    const float aw = alpha_w[0], bw = beta_w[0];
    const float s  = aa * aw;
    const float ca = aa * bw;
    const float cw = aw * ba;
    const float cc = (float)K_DIM * ba * bw;

    #pragma unroll
    for (int fm = 0; fm < 2; ++fm) {
        #pragma unroll
        for (int fn = 0; fn < 2; ++fn) {
            const int n = nBase + wn * 32 + fn * 16 + l16;
            const float swn = sw[n];
            const float bn  = bias[n];
            #pragma unroll
            for (int r = 0; r < 8; ++r) {
                const int m = mBase + wm * 32 + fm * 16 + r + lh * 8;
                out[(size_t)m * N_COLS + n] =
                    s * acc[fm][fn][r] - ca * sx[m] - cw * swn + cc + bn;
            }
        }
    }
}

extern "C" void kernel_launch(void* const* d_in, const int* in_sizes, int n_in,
                              void* d_out, int out_size, void* d_ws, size_t ws_size,
                              hipStream_t stream)
{
    const float* x       = (const float*)d_in[0];
    const float* weight  = (const float*)d_in[1];
    const float* bias    = (const float*)d_in[2];
    const float* alpha_a = (const float*)d_in[3];
    const float* beta_a  = (const float*)d_in[4];
    const float* alpha_w = (const float*)d_in[5];
    const float* beta_w  = (const float*)d_in[6];
    const float* a_w     = (const float*)d_in[7];
    float* out           = (float*)d_out;

    char* ws = (char*)d_ws;
    unsigned char* qx = (unsigned char*)ws;
    unsigned char* qw = (unsigned char*)(ws + (size_t)M_ROWS * K_DIM);
    float* sx = (float*)(ws + (size_t)M_ROWS * K_DIM + (size_t)N_COLS * K_DIM);
    float* sw = sx + M_ROWS;

    fwht_actquant_kernel<<<M_ROWS, 256, 0, stream>>>(x, alpha_a, beta_a, qx, sx);
    wquant_kernel<<<N_COLS, 256, 0, stream>>>(weight, alpha_w, beta_w, a_w, qw, sw);

    dim3 grid(N_COLS / BN, M_ROWS / BM);
    gemm_fp8_kernel<<<grid, 512, 0, stream>>>(qx, qw, bias, sx, sw,
                                              alpha_a, beta_a, alpha_w, beta_w, out);
}
